// ExplicitSpacialEncoding_41257455846144
// MI455X (gfx1250) — compile-verified, matrix operand order checked by reading, not run on hardware
//
#include <hip/hip_runtime.h>
#include <hip/hip_bf16.h>

typedef __attribute__((ext_vector_type(2)))  float  v2f;
typedef __attribute__((ext_vector_type(8)))  float  v8f;
typedef __attribute__((ext_vector_type(16))) __bf16 v16bf;

#define KDIM 64     // P*P = 8*8
#define NJ   9      // spatial-embedding channels
#define INCH 128    // input channels
#define OUTB (INCH * NJ)   // 1152 outputs per batch item

// One block (8 wave32) per batch item b. Wave w computes the 16x16 (padded N)
// output tile for rows i in [16w, 16w+16) via fp32 WMMA over K=64 (16 steps of
// 16x16x4). L2 norm over all 1152 outputs is done block-locally.
__global__ __launch_bounds__(256) void ese_wmma_kernel(
    const float* __restrict__ x,     // (B, 128, 8, 8)
    const float* __restrict__ emb,   // (9, 8, 8)
    float* __restrict__ out)         // (B, 1152)
{
  const int b    = blockIdx.x;
  const int tid  = threadIdx.x;
  const int wave = tid >> 5;      // 0..7 -> M tile
  const int lane = tid & 31;
  const int hi   = lane >> 4;     // half-wave select (K sub-pair / M+8 rows)
  const int lo   = lane & 15;     // M row in A-frag, N column in B/C/D-frag

  const int    row  = wave * 16 + lo;                       // i index
  const float* xrow = x + ((size_t)b * INCH + row) * KDIM;  // 64 floats, contiguous

  const bool  jvalid = (lo < NJ);
  const float jm     = jvalid ? 1.0f : 0.0f;                // zero-pad N to 16
  const float* erow  = emb + (size_t)(jvalid ? lo : 0) * KDIM;

  v8f acc = {0.f, 0.f, 0.f, 0.f, 0.f, 0.f, 0.f, 0.f};

#if defined(__has_builtin) && __has_builtin(__builtin_amdgcn_wmma_f32_16x16x4_f32)
  // Exact-fp32 path: D(16x16) += A(16x4) * B(4x16), 16 K-steps.
  // A layout (ISA 7.12.2): lane lo, VGPR0=K(4k+2hi), VGPR1=K(4k+2hi+1).
  #pragma unroll
  for (int k = 0; k < 16; ++k) {
    const int koff = 4 * k + 2 * hi;
    v2f a = __builtin_nontemporal_load((const v2f*)(xrow + koff)); // streamed once
    v2f bb;
    bb.x = erow[koff]     * jm;   // B(K,N): column N = lo, rows K = koff, koff+1
    bb.y = erow[koff + 1] * jm;
    acc = __builtin_amdgcn_wmma_f32_16x16x4_f32(
        /*neg_a=*/false, a, /*neg_b=*/false, bb,
        /*c_mod=*/(short)0, acc, /*reuse_a=*/false, /*reuse_b=*/false);
  }
#else
  // Fallback (codegen-confirmed builtin): bf16 16x16x32, two K-steps.
  #pragma unroll
  for (int s = 0; s < 2; ++s) {
    v16bf av, bv;
    #pragma unroll
    for (int j = 0; j < 8; ++j) {
      // 16-bit A layout (ISA 7.12.2): VGPR j holds K pair base:
      const int k0 = 32 * s + (j < 4 ? 2 * j : 2 * j + 8) + 8 * hi;
      av[2 * j]     = (__bf16)xrow[k0];
      av[2 * j + 1] = (__bf16)xrow[k0 + 1];
      bv[2 * j]     = (__bf16)(erow[k0]     * jm);
      bv[2 * j + 1] = (__bf16)(erow[k0 + 1] * jm);
    }
    acc = __builtin_amdgcn_wmma_f32_16x16x32_bf16(
        false, av, false, bv, (short)0, acc, false, false);
  }
#endif

  // ---- block-local L2 norm over all 1152 outputs of this batch item ----
  // Columns N>=9 are exactly zero (B padded), so no masking needed.
  float s2 = 0.f;
  #pragma unroll
  for (int p = 0; p < 8; ++p) s2 += acc[p] * acc[p];
  #pragma unroll
  for (int off = 16; off > 0; off >>= 1) s2 += __shfl_xor(s2, off, 32);

  __shared__ float wsum[8];
  if (lane == 0) wsum[wave] = s2;
  __syncthreads();
  float tot = 0.f;
  #pragma unroll
  for (int w = 0; w < 8; ++w) tot += wsum[w];
  const float inv = rsqrtf(tot + 1e-10f);

  // C/D layout: lane (hi, lo), VGPR p -> element (M = 16*wave + p + 8*hi, N = lo)
  if (jvalid) {
    float* op = out + (size_t)b * OUTB;
    #pragma unroll
    for (int p = 0; p < 8; ++p) {
      const int m = wave * 16 + p + 8 * hi;
      __builtin_nontemporal_store(acc[p] * inv, op + m * NJ + lo);
    }
  }
}

extern "C" void kernel_launch(void* const* d_in, const int* in_sizes, int n_in,
                              void* d_out, int out_size, void* d_ws, size_t ws_size,
                              hipStream_t stream) {
  const float* x   = (const float*)d_in[0];   // (B,128,8,8) fp32
  const float* emb = (const float*)d_in[1];   // (9,8,8) fp32
  float* out = (float*)d_out;                 // (B,1152) fp32

  const int Bn = in_sizes[0] / (INCH * KDIM); // 4096
  ese_wmma_kernel<<<dim3(Bn), dim3(256), 0, stream>>>(x, emb, out);
}
